// QuantLinear_40080634806821
// MI455X (gfx1250) — compile-verified, matrix operand order checked by reading, not run on hardware
//
#include <hip/hip_runtime.h>

// ---------------------------------------------------------------------------
// QuIP#-style quantized linear for MI455X (gfx1250, wave32, WMMA).
//   out = ( FWHT( FWHT(x*SU)*(Wscale/64) @ W^T ) / 64 ) * SV + bias
//   W = cb[Qidxs]  (256x4 f32 codebook, 8-bit indices per 4-wide group)
// GEMM in bf16 via v_wmma_f32_16x16x32_bf16; double-buffered LDS, BK=64.
// ---------------------------------------------------------------------------

#define IN_F   4096
#define OUT_F  4096
#define NTOK   8192          // 4 * 2048 tokens

typedef __attribute__((ext_vector_type(16))) __bf16 v16bf;
typedef __attribute__((ext_vector_type(8)))  float  v8f;

__device__ __forceinline__ unsigned short f2bf(float f) {
    union { float f; unsigned int u; } v; v.f = f;
    unsigned int r = v.u + 0x7FFFu + ((v.u >> 16) & 1u);   // round-to-nearest-even
    return (unsigned short)(r >> 16);
}
__device__ __forceinline__ unsigned int pack2bf(float lo, float hi) {
    return (unsigned int)f2bf(lo) | ((unsigned int)f2bf(hi) << 16);
}

// ---------------------------------------------------------------------------
// Kernel 1: decode codebook weights -> bf16 W[o][i], row-major over i.
// ---------------------------------------------------------------------------
__global__ __launch_bounds__(256) void decode_kernel(
    const int* __restrict__ Qidxs, const float4* __restrict__ cb,
    uint2* __restrict__ Wbf)
{
    const int gid = blockIdx.x * 256 + threadIdx.x;   // 0 .. OUT_F*IN_F/4-1
    const int idx = Qidxs[gid];
    const float4 c = cb[idx];
    Wbf[gid] = make_uint2(pack2bf(c.x, c.y), pack2bf(c.z, c.w));
}

// ---------------------------------------------------------------------------
// Kernel 2: x = FWHT(input*SU) * (Wscale/64), packed to bf16.
// ---------------------------------------------------------------------------
__global__ __launch_bounds__(256) void had_in_kernel(
    const float* __restrict__ input, const float* __restrict__ SU,
    const float* __restrict__ Wscale, unsigned short* __restrict__ Xbf)
{
    __shared__ float lds[IN_F];
    const int row = blockIdx.x;
    const int tid = threadIdx.x;

    const float4* in4 = (const float4*)(input + (size_t)row * IN_F);
    const float4* su4 = (const float4*)SU;
    for (int i = tid; i < IN_F / 4; i += 256) {
        float4 a = in4[i], s = su4[i];
        lds[4*i+0] = a.x * s.x; lds[4*i+1] = a.y * s.y;
        lds[4*i+2] = a.z * s.z; lds[4*i+3] = a.w * s.w;
    }
    for (int h = 1; h < IN_F; h <<= 1) {
        __syncthreads();
        for (int p = tid; p < IN_F / 2; p += 256) {
            int base = ((p / h) * (h << 1)) + (p % h);
            float a = lds[base], b = lds[base + h];
            lds[base] = a + b; lds[base + h] = a - b;
        }
    }
    __syncthreads();
    const float sc = Wscale[0] * 0.015625f;   // 1/sqrt(4096) = 1/64
    unsigned int* X32 = (unsigned int*)(Xbf + (size_t)row * IN_F);
    for (int i = tid; i < IN_F / 2; i += 256)
        X32[i] = pack2bf(lds[2*i] * sc, lds[2*i+1] * sc);
}

// ---------------------------------------------------------------------------
// Kernel 3: out[t][o] = sum_i Xbf[t][i] * Wbf[o][i]   (bf16 WMMA, f32 accum)
// Block tile 128(M) x 128(N), K-tile 64 (two WMMA k-steps per barrier).
// 8 wave32s in a 4x2 grid; each wave owns 32x64 = 2x4 fragments of 16x16.
// Double-buffered LDS: one barrier per K-tile (16 WMMAs / barrier).
// ---------------------------------------------------------------------------
__global__ __launch_bounds__(256) void gemm_kernel(
    const unsigned short* __restrict__ Xbf,
    const unsigned short* __restrict__ Wbf,
    float* __restrict__ out)
{
    __shared__ unsigned short Ash[2][128 * 64];   // [buf][m][k], k contiguous
    __shared__ unsigned short Bsh[2][128 * 64];   // [buf][n][k], k contiguous

    const int tid  = threadIdx.x;
    const int lane = tid & 31;
    const int w    = tid >> 5;
    const int wm   = w >> 1;          // 0..3  -> m offset wm*32
    const int wn   = w & 1;           // 0..1  -> n offset wn*64
    const int bm   = blockIdx.y * 128;
    const int bn   = blockIdx.x * 128;

    // Global load map: tile is 128 rows x 64 elems (128 B/row).
    // Thread owns 4 x 16B segments: rows r, r+32, r+64, r+96 at elem offset sg.
    const int r  = tid >> 3;                  // 0..31
    const int sg = (tid & 7) << 3;            // 0,8,..,56
    const unsigned short* Ag = Xbf + (size_t)(bm + r) * IN_F + sg;
    const unsigned short* Bg = Wbf + (size_t)(bn + r) * IN_F + sg;
    const size_t rowStep = (size_t)32 * IN_F;

    // Fragment read map (contiguous k order, same for A and B): lane l holds row (l&15),
    // 16 consecutive K starting at (l>>4)*16 -> one 32B LDS read per fragment.
    const int aoff = (wm * 32 + (lane & 15)) * 64 + (lane >> 4) * 16;
    const int boff = (wn * 64 + (lane & 15)) * 64 + (lane >> 4) * 16;

    v8f acc[2][4];
    const v8f vzero = {0.f,0.f,0.f,0.f,0.f,0.f,0.f,0.f};
    #pragma unroll
    for (int mi = 0; mi < 2; ++mi)
        #pragma unroll
        for (int ni = 0; ni < 4; ++ni)
            acc[mi][ni] = vzero;

    // Prime: K-tile 0 -> regs -> LDS buffer 0.
    uint4 arg[4], brg[4];
    #pragma unroll
    for (int j = 0; j < 4; ++j) {
        arg[j] = *(const uint4*)(Ag + j * rowStep);
        brg[j] = *(const uint4*)(Bg + j * rowStep);
    }
    #pragma unroll
    for (int j = 0; j < 4; ++j) {
        *(uint4*)&Ash[0][(r + 32 * j) * 64 + sg] = arg[j];
        *(uint4*)&Bsh[0][(r + 32 * j) * 64 + sg] = brg[j];
    }
    __syncthreads();

    const int KT = IN_F / 64;   // 64 K-tiles
    int ping = 0;
    for (int kt = 0; kt < KT; ++kt) {
        if (kt + 1 < KT) {      // fetch K-tile kt+1 into regs while computing
            const int ko = (kt + 1) * 64;
            #pragma unroll
            for (int j = 0; j < 4; ++j) {
                arg[j] = *(const uint4*)(Ag + ko + j * rowStep);
                brg[j] = *(const uint4*)(Bg + ko + j * rowStep);
            }
        }
        if (kt + 2 < KT) {      // warm L2 for K-tile kt+2 (global_prefetch)
            __builtin_prefetch(Ag + (kt + 2) * 64, 0, 1);
            __builtin_prefetch(Bg + (kt + 2) * 64, 0, 1);
        }

        const unsigned short* As = &Ash[ping][0];
        const unsigned short* Bs = &Bsh[ping][0];
        #pragma unroll
        for (int ks = 0; ks < 2; ++ks) {
            v16bf a0 = *(const v16bf*)(As + aoff +        ks * 32);
            v16bf a1 = *(const v16bf*)(As + aoff + 1024 + ks * 32);
            v16bf b0 = *(const v16bf*)(Bs + boff +        ks * 32);
            v16bf b1 = *(const v16bf*)(Bs + boff + 1024 + ks * 32);
            v16bf b2 = *(const v16bf*)(Bs + boff + 2048 + ks * 32);
            v16bf b3 = *(const v16bf*)(Bs + boff + 3072 + ks * 32);

            acc[0][0] = __builtin_amdgcn_wmma_f32_16x16x32_bf16(false, a0, false, b0, (short)0, acc[0][0], false, false);
            acc[0][1] = __builtin_amdgcn_wmma_f32_16x16x32_bf16(false, a0, false, b1, (short)0, acc[0][1], false, false);
            acc[0][2] = __builtin_amdgcn_wmma_f32_16x16x32_bf16(false, a0, false, b2, (short)0, acc[0][2], false, false);
            acc[0][3] = __builtin_amdgcn_wmma_f32_16x16x32_bf16(false, a0, false, b3, (short)0, acc[0][3], false, false);
            acc[1][0] = __builtin_amdgcn_wmma_f32_16x16x32_bf16(false, a1, false, b0, (short)0, acc[1][0], false, false);
            acc[1][1] = __builtin_amdgcn_wmma_f32_16x16x32_bf16(false, a1, false, b1, (short)0, acc[1][1], false, false);
            acc[1][2] = __builtin_amdgcn_wmma_f32_16x16x32_bf16(false, a1, false, b2, (short)0, acc[1][2], false, false);
            acc[1][3] = __builtin_amdgcn_wmma_f32_16x16x32_bf16(false, a1, false, b3, (short)0, acc[1][3], false, false);
        }

        if (kt + 1 < KT) {      // store K-tile kt+1 into the other buffer
            const int pong = ping ^ 1;
            #pragma unroll
            for (int j = 0; j < 4; ++j) {
                *(uint4*)&Ash[pong][(r + 32 * j) * 64 + sg] = arg[j];
                *(uint4*)&Bsh[pong][(r + 32 * j) * 64 + sg] = brg[j];
            }
        }
        __syncthreads();        // single barrier per K-tile
        ping ^= 1;
    }

    // C/D layout: VGPR r, lane l -> row = r + 8*(l>>4), col = l&15.
    const int mrow = bm + wm * 32 + (lane >> 4) * 8;
    const int ncol = bn + wn * 64 + (lane & 15);
    #pragma unroll
    for (int mi = 0; mi < 2; ++mi)
        #pragma unroll
        for (int ni = 0; ni < 4; ++ni)
            #pragma unroll
            for (int rr = 0; rr < 8; ++rr)
                out[(size_t)(mrow + mi * 16 + rr) * OUT_F + (ncol + ni * 16)] = acc[mi][ni][rr];
}

// ---------------------------------------------------------------------------
// Kernel 4: out_row = FWHT(out_row)/64 * SV + bias   (in-place on d_out)
// ---------------------------------------------------------------------------
__global__ __launch_bounds__(256) void had_out_kernel(
    float* __restrict__ out, const float* __restrict__ SV,
    const float* __restrict__ bias)
{
    __shared__ float lds[OUT_F];
    const int row = blockIdx.x;
    const int tid = threadIdx.x;

    float4* o4 = (float4*)(out + (size_t)row * OUT_F);
    for (int i = tid; i < OUT_F / 4; i += 256) {
        float4 v = o4[i];
        lds[4*i+0] = v.x; lds[4*i+1] = v.y; lds[4*i+2] = v.z; lds[4*i+3] = v.w;
    }
    for (int h = 1; h < OUT_F; h <<= 1) {
        __syncthreads();
        for (int p = tid; p < OUT_F / 2; p += 256) {
            int base = ((p / h) * (h << 1)) + (p % h);
            float a = lds[base], b = lds[base + h];
            lds[base] = a + b; lds[base + h] = a - b;
        }
    }
    __syncthreads();
    const float sc = 0.015625f;   // 1/sqrt(4096)
    const float4* sv4 = (const float4*)SV;
    const float4* bi4 = (const float4*)bias;
    for (int i = tid; i < OUT_F / 4; i += 256) {
        float4 s = sv4[i], b = bi4[i], r;
        r.x = lds[4*i+0] * sc * s.x + b.x;
        r.y = lds[4*i+1] * sc * s.y + b.y;
        r.z = lds[4*i+2] * sc * s.z + b.z;
        r.w = lds[4*i+3] * sc * s.w + b.w;
        o4[i] = r;
    }
}

// ---------------------------------------------------------------------------
extern "C" void kernel_launch(void* const* d_in, const int* in_sizes, int n_in,
                              void* d_out, int out_size, void* d_ws, size_t ws_size,
                              hipStream_t stream) {
    (void)in_sizes; (void)n_in; (void)out_size; (void)ws_size;
    const float* input  = (const float*)d_in[0];
    const float* SU     = (const float*)d_in[1];
    const float* SV     = (const float*)d_in[2];
    const float* cb     = (const float*)d_in[3];
    const int*   Qidxs  = (const int*)  d_in[4];
    const float* Wscale = (const float*)d_in[5];
    const float* bias   = (const float*)d_in[6];
    float*       out    = (float*)d_out;

    // Workspace: Xbf (8192x4096 bf16 = 64MB) @ 0, Wbf (4096x4096 bf16 = 32MB) after.
    unsigned short* Xbf = (unsigned short*)d_ws;
    unsigned short* Wbf = (unsigned short*)((char*)d_ws + (size_t)NTOK * IN_F * 2);

    decode_kernel<<<(OUT_F * (IN_F / 4)) / 256, 256, 0, stream>>>(
        Qidxs, (const float4*)cb, (uint2*)Wbf);
    had_in_kernel<<<NTOK, 256, 0, stream>>>(input, SU, Wscale, Xbf);
    gemm_kernel<<<dim3(OUT_F / 128, NTOK / 128), 256, 0, stream>>>(Xbf, Wbf, out);
    had_out_kernel<<<NTOK, 256, 0, stream>>>(out, SV, bias);
}
